// IntersectionNeighborMixer_19610820674005
// MI455X (gfx1250) — matrix operand order repaired and checked by reading, not compiled, not run on hardware
//
#include <hip/hip_runtime.h>

typedef __attribute__((ext_vector_type(16))) _Float16 v16h;
typedef __attribute__((ext_vector_type(8)))  _Float16 v8h;
typedef __attribute__((ext_vector_type(8)))  float    v8f;

#define N_NODES 100000
#define N_EDGES 1600000
#define D_EMB   128
#define H_DIM   256
#define TWO_D   256
#define LSTR    264   // LDS row stride in f16 elems: 256 + 8 pad (bank spread, keeps 16B align)

// ---------------------------------------------------------------------------
// Kernel 1: zero the accumulation workspace (harness poisons ws with 0xAA)
// ---------------------------------------------------------------------------
__global__ void k_zero(float* __restrict__ nbr_sum, float* __restrict__ counts) {
    int i = blockIdx.x * 256 + threadIdx.x;          // grid covers N_NODES*D exactly
    nbr_sum[i] = 0.0f;
    if (i < N_NODES) counts[i] = 0.0f;
}

// ---------------------------------------------------------------------------
// Kernel 2: convert + transpose weights to f16, row-major over K.
//   W1T[n][k] = W1[k][n]   (n,k in [0,256))
//   W2T[n][k] = W2[k][n]   (n in [0,128), k in [0,256))
// ---------------------------------------------------------------------------
__global__ void k_wconv(const float* __restrict__ W1, const float* __restrict__ W2,
                        _Float16* __restrict__ W1T, _Float16* __restrict__ W2T) {
    int i = blockIdx.x * 256 + threadIdx.x;
    if (i < H_DIM * TWO_D) {
        int n = i >> 8, k = i & 255;
        W1T[n * TWO_D + k] = (_Float16)W1[k * H_DIM + n];
    } else {
        int j = i - H_DIM * TWO_D;
        if (j < D_EMB * H_DIM) {
            int n = j >> 8, k = j & 255;
            W2T[n * H_DIM + k] = (_Float16)W2[k * D_EMB + n];
        }
    }
}

// ---------------------------------------------------------------------------
// Kernel 3: edge scatter. One wave32 per edge: float4 gather per lane (512B
// row fully coalesced) + 4 hardware f32 atomics into nbr_sum (L2-resident).
// ---------------------------------------------------------------------------
__global__ void k_scatter(const float* __restrict__ emb,
                          const int* __restrict__ esrc, const int* __restrict__ edst,
                          float* __restrict__ nbr_sum, float* __restrict__ counts) {
    int t = blockIdx.x * 256 + threadIdx.x;
    int e = t >> 5;
    int lane = t & 31;
    if (e >= N_EDGES) return;
    int s = esrc[e];
    int d = edst[e];
    const float4 v = *(const float4*)(emb + (size_t)s * D_EMB + lane * 4);
    float* dst = nbr_sum + (size_t)d * D_EMB + lane * 4;
    unsafeAtomicAdd(dst + 0, v.x);
    unsafeAtomicAdd(dst + 1, v.y);
    unsafeAtomicAdd(dst + 2, v.z);
    unsafeAtomicAdd(dst + 3, v.w);
    if (lane == 0) unsafeAtomicAdd(counts + d, 1.0f);
}

// ---------------------------------------------------------------------------
// Kernel 4: fused mean + concat + MLP for a 16-node tile.
//   Stage A: concat[16][256] (f16) into LDS
//   Stage B: h = relu(concat @ W1 + b1), WMMA f16->f32, h (f16) into LDS
//   Stage C: out = h @ W2 + b2, WMMA f16->f32, f32 store
// ---------------------------------------------------------------------------
__global__ __launch_bounds__(256) void k_mlp(
    const float* __restrict__ emb, const float* __restrict__ nbr_sum,
    const float* __restrict__ counts,
    const _Float16* __restrict__ W1T, const float* __restrict__ bias1,
    const _Float16* __restrict__ W2T, const float* __restrict__ bias2,
    float* __restrict__ out) {
    __shared__ _Float16 sA[16 * LSTR];   // concat tile
    __shared__ _Float16 sH[16 * LSTR];   // hidden tile

    const int node0 = blockIdx.x * 16;
    const int t = threadIdx.x;

    // ---- Stage A: build concat tile in LDS (f16) ----
    #pragma unroll
    for (int j = 0; j < 16; ++j) {
        int idx = t + 256 * j;            // [0, 4096)
        int row = idx >> 8;
        int col = idx & 255;
        int node = node0 + row;
        float v;
        if (col < D_EMB) {
            v = emb[(size_t)node * D_EMB + col];
        } else {
            float c = counts[node];
            float s = nbr_sum[(size_t)node * D_EMB + (col - D_EMB)];
            float e = emb[(size_t)node * D_EMB + (col - D_EMB)];
            v = (c > 0.0f) ? (s / c) : e;   // segment mean, fallback to own emb
        }
        sA[row * LSTR + col] = (_Float16)v;
    }
    __syncthreads();

    const int wave  = t >> 5;
    const int lane  = t & 31;
    const int lrow  = lane & 15;      // row (A) or col (B, C/D) within 16-tile
    const int khalf = lane >> 4;      // 0 or 1
    const int akb   = khalf * 16;     // A-frag K sub-base (contiguous layout)
    const int bkb   = khalf * 16;     // B-frag K sub-base (contiguous layout)

    // ---- Stage B: GEMM1, wave w owns hidden cols [32w, 32w+32) ----
    v8f c0 = {}; v8f c1 = {};
    const int n0 = wave * 32 + lrow;
    const int n1 = wave * 32 + 16 + lrow;
    for (int k0 = 0; k0 < TWO_D; k0 += 32) {
        v16h a;
        {
            const v8h lo = *(const v8h*)(&sA[lrow * LSTR + k0 + akb]);
            const v8h hi = *(const v8h*)(&sA[lrow * LSTR + k0 + akb + 8]);
            #pragma unroll
            for (int i = 0; i < 8; ++i) { a[i] = lo[i]; a[i + 8] = hi[i]; }
        }
        v16h b0, b1v;
        {
            const v8h x = *(const v8h*)(&W1T[n0 * TWO_D + k0 + bkb]);
            const v8h y = *(const v8h*)(&W1T[n0 * TWO_D + k0 + bkb + 8]);
            #pragma unroll
            for (int i = 0; i < 8; ++i) { b0[i] = x[i]; b0[i + 8] = y[i]; }
        }
        {
            const v8h x = *(const v8h*)(&W1T[n1 * TWO_D + k0 + bkb]);
            const v8h y = *(const v8h*)(&W1T[n1 * TWO_D + k0 + bkb + 8]);
            #pragma unroll
            for (int i = 0; i < 8; ++i) { b1v[i] = x[i]; b1v[i + 8] = y[i]; }
        }
        c0 = __builtin_amdgcn_wmma_f32_16x16x32_f16(false, a, false, b0,  (short)0, c0, false, false);
        c1 = __builtin_amdgcn_wmma_f32_16x16x32_f16(false, a, false, b1v, (short)0, c1, false, false);
    }
    {   // bias + relu, stash h as f16 in LDS
        const float bb0 = bias1[n0];
        const float bb1 = bias1[n1];
        #pragma unroll
        for (int i = 0; i < 8; ++i) {
            int row = i + khalf * 8;           // C/D layout: VGPR i, lane-half selects M/M+8
            float h0 = c0[i] + bb0; h0 = h0 > 0.0f ? h0 : 0.0f;
            float h1 = c1[i] + bb1; h1 = h1 > 0.0f ? h1 : 0.0f;
            sH[row * LSTR + n0] = (_Float16)h0;
            sH[row * LSTR + n1] = (_Float16)h1;
        }
    }
    __syncthreads();

    // ---- Stage C: GEMM2, wave w owns out cols [16w, 16w+16) ----
    v8f c2 = {};
    const int n2 = wave * 16 + lrow;
    for (int k0 = 0; k0 < H_DIM; k0 += 32) {
        v16h a;
        {
            const v8h lo = *(const v8h*)(&sH[lrow * LSTR + k0 + akb]);
            const v8h hi = *(const v8h*)(&sH[lrow * LSTR + k0 + akb + 8]);
            #pragma unroll
            for (int i = 0; i < 8; ++i) { a[i] = lo[i]; a[i + 8] = hi[i]; }
        }
        v16h b;
        {
            const v8h x = *(const v8h*)(&W2T[n2 * H_DIM + k0 + bkb]);
            const v8h y = *(const v8h*)(&W2T[n2 * H_DIM + k0 + bkb + 8]);
            #pragma unroll
            for (int i = 0; i < 8; ++i) { b[i] = x[i]; b[i + 8] = y[i]; }
        }
        c2 = __builtin_amdgcn_wmma_f32_16x16x32_f16(false, a, false, b, (short)0, c2, false, false);
    }
    {
        const float bb = bias2[n2];
        #pragma unroll
        for (int i = 0; i < 8; ++i) {
            int row = i + khalf * 8;
            out[(size_t)(node0 + row) * D_EMB + n2] = c2[i] + bb;
        }
    }
}

// ---------------------------------------------------------------------------
extern "C" void kernel_launch(void* const* d_in, const int* in_sizes, int n_in,
                              void* d_out, int out_size, void* d_ws, size_t ws_size,
                              hipStream_t stream) {
    const float* emb  = (const float*)d_in[0];
    const int*   esrc = (const int*)d_in[1];
    const int*   edst = (const int*)d_in[2];
    const float* W1   = (const float*)d_in[3];
    const float* b1   = (const float*)d_in[4];
    const float* W2   = (const float*)d_in[5];
    const float* b2   = (const float*)d_in[6];
    float*       out  = (float*)d_out;

    // workspace layout (bytes): nbr_sum | counts | W1T(f16) | W2T(f16)
    char* ws = (char*)d_ws;
    float*    nbr_sum = (float*)ws;                                           // N*D f32 = 51.2 MB
    float*    counts  = (float*)(ws + (size_t)N_NODES * D_EMB * 4);           // N f32
    _Float16* W1T     = (_Float16*)(ws + (size_t)N_NODES * D_EMB * 4
                                       + (size_t)N_NODES * 4);                // 128 KB
    _Float16* W2T     = W1T + H_DIM * TWO_D;                                  // 64 KB

    k_zero   <<<(N_NODES * D_EMB) / 256, 256, 0, stream>>>(nbr_sum, counts);
    k_wconv  <<<(H_DIM * TWO_D + D_EMB * H_DIM) / 256, 256, 0, stream>>>(W1, W2, W1T, W2T);
    k_scatter<<<(N_EDGES * 32) / 256, 256, 0, stream>>>(emb, esrc, edst, nbr_sum, counts);
    k_mlp    <<<N_NODES / 16, 256, 0, stream>>>(emb, nbr_sum, counts, W1T, b1, W2T, b2, out);
}
